// TransformerEncoderBlock_38216618999837
// MI455X (gfx1250) — matrix operand order repaired and checked by reading, not compiled, not run on hardware
//
#include <hip/hip_runtime.h>
#include <cstdint>
#include <cstddef>

typedef __bf16 bf16;
typedef __bf16 v16bf  __attribute__((ext_vector_type(16)));
typedef __bf16 bf16x8 __attribute__((ext_vector_type(8)));
typedef float  v8f    __attribute__((ext_vector_type(8)));
typedef int    v4i    __attribute__((ext_vector_type(4)));

#define B_   8
#define S_   2048
#define E_   768
#define H_   3072
#define NTOK (B_ * S_)                  /* 16384 */
#define ATTN_SCALE 0.03608439182435161f /* 1/sqrt(768) */

static constexpr int KP = 40; /* padded LDS row (bf16 elems) -> 80B, 16B-aligned */

static __device__ inline v16bf load_frag(const bf16* p0, const bf16* p1) {
  v16bf r;
  *(bf16x8*)&r       = *(const bf16x8*)p0;
  *((bf16x8*)&r + 1) = *(const bf16x8*)p1;
  return r;
}

/* ---- CDNA5 async global->LDS copy (ASYNCcnt-tracked, bypasses VGPRs) ----
   clang prototype: (int4 addrspace(1)*, int4 addrspace(3)*, imm offset, imm cpol) */

typedef __attribute__((address_space(1))) v4i* glb_v4i_ptr;
typedef __attribute__((address_space(3))) v4i* lds_v4i_ptr;

#if __has_builtin(__builtin_amdgcn_global_load_async_to_lds_b128)
static __device__ inline void async_copy_b128(void* lds_dst, const void* g_src) {
  __builtin_amdgcn_global_load_async_to_lds_b128(
      (glb_v4i_ptr)g_src, (lds_v4i_ptr)lds_dst, 0, 0);
}
#else
static __device__ inline void async_copy_b128(void* lds_dst, const void* g_src) {
  uint32_t loff = (uint32_t)(uintptr_t)(lds_v4i_ptr)lds_dst;
  uint64_t ga   = (uint64_t)(uintptr_t)(glb_v4i_ptr)g_src;
  asm volatile("global_load_async_to_lds_b128 %0, %1, off" :: "v"(loff), "v"(ga) : "memory");
}
#endif

#if __has_builtin(__builtin_amdgcn_s_wait_asynccnt)
#define WAIT_ASYNCCNT(n) __builtin_amdgcn_s_wait_asynccnt(n)
#else
#define WAIT_ASYNCCNT(n) asm volatile("s_wait_asynccnt %0" :: "i"(n) : "memory")
#endif

/* ---------------- elementwise conversions / transposes ---------------- */

__global__ void conv_f32_bf16(const float* __restrict__ s, bf16* __restrict__ d, int n) {
  int i = blockIdx.x * blockDim.x + threadIdx.x;
  if (i < n) d[i] = (bf16)s[i];
}

/* dst[N][K] = (bf16) src[K][N] */
__global__ void transpose_conv(const float* __restrict__ s, bf16* __restrict__ d, int K, int N) {
  int i = blockIdx.x * blockDim.x + threadIdx.x;
  if (i >= K * N) return;
  int k = i % K, n = i / K;
  d[(size_t)n * K + k] = (bf16)s[(size_t)k * N + n];
}

/* vT[b][e][t] = v[b][t][e] */
__global__ void transpose_v(const bf16* __restrict__ v, bf16* __restrict__ vt) {
  int i = blockIdx.x * blockDim.x + threadIdx.x;
  if (i >= B_ * E_ * S_) return;
  int t = i % S_;
  int e = (i / S_) % E_;
  int b = i / (S_ * E_);
  vt[i] = v[((size_t)(b * S_ + t)) * E_ + e];
}

/* ---------------- WMMA GEMM: C = A[M,K] x Bt[N,K]^T + bias ----------------
   128x128 block tile, 8 waves, 64x32 per wave, K-step 32.
   Double-buffered async global->LDS staging overlapped with WMMA. */

__global__ __launch_bounds__(256) void gemm_bf16_wmma(
    const bf16* __restrict__ A, const bf16* __restrict__ Bt,
    const float* __restrict__ bias, float* __restrict__ outf,
    bf16* __restrict__ outb, int M, int N, int K, int relu)
{
  __shared__ bf16 la[2][128 * KP];
  __shared__ bf16 lb[2][128 * KP];

  const int t    = threadIdx.x;
  const int w    = t >> 5;
  const int lane = t & 31;
  const int half = lane >> 4;
  const int ln   = lane & 15;
  const int wr   = w >> 2;   /* 0..1 : M offset wr*64 */
  const int wc   = w & 3;    /* 0..3 : N offset wc*32 */
  const int m0   = blockIdx.y * 128;
  const int n0   = blockIdx.x * 128;

  v8f acc[4][2];
#pragma unroll
  for (int mt = 0; mt < 4; mt++)
#pragma unroll
    for (int nt = 0; nt < 2; nt++)
      acc[mt][nt] = (v8f){0.f, 0.f, 0.f, 0.f, 0.f, 0.f, 0.f, 0.f};

  /* 512 16-byte chunks per tile; 2 A + 2 B async instructions per thread */
  auto stage_async = [&](int buf, int k0) {
#pragma unroll
    for (int c = t; c < 512; c += 256) {
      int row = c >> 2, seg = (c & 3) * 8;
      async_copy_b128(&la[buf][row * KP + seg], &A[(size_t)(m0 + row) * K + k0 + seg]);
      async_copy_b128(&lb[buf][row * KP + seg], &Bt[(size_t)(n0 + row) * K + k0 + seg]);
    }
  };

  stage_async(0, 0);
  int cur = 0;
  for (int k0 = 0; k0 < K; k0 += 32) {
    const bool pf = (k0 + 32) < K;
    if (pf) stage_async(cur ^ 1, k0 + 32); /* prefetch next tile into other buffer */
    if (pf) { WAIT_ASYNCCNT(4); }          /* in-order: <=4 left => current tile landed */
    else    { WAIT_ASYNCCNT(0); }
    __syncthreads();                       /* all waves waited their own ASYNCcnt */

    v16bf af[4], bfv[2];
#pragma unroll
    for (int mt = 0; mt < 4; mt++) {
      int r = (wr * 64 + mt * 16 + ln) * KP;
      af[mt] = load_frag(&la[cur][r + 8 * half], &la[cur][r + 16 + 8 * half]);
    }
#pragma unroll
    for (int nt = 0; nt < 2; nt++) {
      int r = (wc * 32 + nt * 16 + ln) * KP;
      bfv[nt] = load_frag(&lb[cur][r + 8 * half], &lb[cur][r + 16 + 8 * half]);
    }
#pragma unroll
    for (int mt = 0; mt < 4; mt++)
#pragma unroll
      for (int nt = 0; nt < 2; nt++)
        acc[mt][nt] = __builtin_amdgcn_wmma_f32_16x16x32_bf16(
            false, af[mt], false, bfv[nt], (short)0, acc[mt][nt], false, false);

    __syncthreads(); /* all waves done reading 'cur' before it is refilled */
    cur ^= 1;
  }

#pragma unroll
  for (int mt = 0; mt < 4; mt++) {
    int rbase = m0 + wr * 64 + mt * 16 + 8 * half;
#pragma unroll
    for (int nt = 0; nt < 2; nt++) {
      int col  = n0 + wc * 32 + nt * 16 + ln;
      float bv = bias[col];
      v8f c    = acc[mt][nt];
#pragma unroll
      for (int v = 0; v < 8; v++) {
        float val = c[v] + bv;
        if (relu) val = fmaxf(val, 0.0f);
        size_t idx = (size_t)(rbase + v) * (size_t)N + col;
        if (outb) outb[idx] = (bf16)val;
        else      outf[idx] = val;
      }
    }
  }
}

/* ---------------- fused attention (flash-style online softmax) ----------------
   scores = K·Qᵀ (reference quirk), softmax over t, out = P·V.
   block = (batch, 16 s-rows); 4 waves. Wave w computes the PARTIAL score over
   its E-slice [w*192, w*192+192); partials are reduced through LDS so QK work
   is not duplicated. Each wave then owns the P·V accumulation for its slice. */

__global__ __launch_bounds__(128) void attn_wmma(
    const bf16* __restrict__ qb, const bf16* __restrict__ kb,
    const bf16* __restrict__ vT, float* __restrict__ h)
{
  __shared__ float sred[4 * 2 * 16 * 16]; /* [wave][frag][row][col] partial S */
  __shared__ bf16  lp[4][16 * KP];        /* per-wave P tile, [s-row][t-local] */

  const int t    = threadIdx.x;
  const int w    = t >> 5;
  const int lane = t & 31;
  const int half = lane >> 4;
  const int ln   = lane & 15;
  const int b    = blockIdx.x >> 7;       /* 128 row-blocks per batch */
  const int s0   = (blockIdx.x & 127) * 16;
  const int ew   = w * 192;

  v8f o[12];
#pragma unroll
  for (int nt = 0; nt < 12; nt++) o[nt] = (v8f){0.f, 0.f, 0.f, 0.f, 0.f, 0.f, 0.f, 0.f};
  float m_run[8], l_run[8];
#pragma unroll
  for (int v = 0; v < 8; v++) { m_run[v] = -3.0e38f; l_run[v] = 0.f; }

  const bf16* krow = kb + ((size_t)(b * S_ + s0 + ln)) * E_; /* A-frag row */

#pragma unroll 1
  for (int t0 = 0; t0 < S_; t0 += 32) {
    /* ---- partial S = K·Qᵀ over this wave's E slice (2 t-tiles) ---- */
    v8f sf0 = (v8f){0.f, 0.f, 0.f, 0.f, 0.f, 0.f, 0.f, 0.f};
    v8f sf1 = sf0;
#pragma unroll
    for (int ee = 0; ee < 6; ee++) {
      int e0 = ew + ee * 32;
      v16bf Af = load_frag(krow + e0 + 8 * half, krow + e0 + 16 + 8 * half);
      const bf16* q0 = qb + ((size_t)(b * S_ + t0 + ln)) * E_ + e0 + 8 * half;
      const bf16* q1 = qb + ((size_t)(b * S_ + t0 + 16 + ln)) * E_ + e0 + 8 * half;
      v16bf B0 = load_frag(q0, q0 + 16);
      v16bf B1 = load_frag(q1, q1 + 16);
      sf0 = __builtin_amdgcn_wmma_f32_16x16x32_bf16(false, Af, false, B0, (short)0, sf0, false, false);
      sf1 = __builtin_amdgcn_wmma_f32_16x16x32_bf16(false, Af, false, B1, (short)0, sf1, false, false);
    }

    /* ---- reduce partial S across waves through LDS ---- */
#pragma unroll
    for (int v = 0; v < 8; v++) {
      int r = v + 8 * half;
      sred[((w * 2 + 0) * 16 + r) * 16 + ln] = sf0[v];
      sred[((w * 2 + 1) * 16 + r) * 16 + ln] = sf1[v];
    }
    __syncthreads();

    /* ---- full S; online softmax (rows live in 16-lane half-groups) ---- */
    float p0[8], p1[8], alpha[8];
#pragma unroll
    for (int v = 0; v < 8; v++) {
      int r = v + 8 * half;
      float a0 = 0.f, a1 = 0.f;
#pragma unroll
      for (int ww = 0; ww < 4; ww++) {
        a0 += sred[((ww * 2 + 0) * 16 + r) * 16 + ln];
        a1 += sred[((ww * 2 + 1) * 16 + r) * 16 + ln];
      }
      a0 *= ATTN_SCALE;
      a1 *= ATTN_SCALE;
      float mx = fmaxf(a0, a1);
      mx = fmaxf(mx, __shfl_xor(mx, 1));
      mx = fmaxf(mx, __shfl_xor(mx, 2));
      mx = fmaxf(mx, __shfl_xor(mx, 4));
      mx = fmaxf(mx, __shfl_xor(mx, 8));
      float mn = fmaxf(m_run[v], mx);
      alpha[v] = __expf(m_run[v] - mn);
      float e0v = __expf(a0 - mn);
      float e1v = __expf(a1 - mn);
      float rs  = e0v + e1v;
      rs += __shfl_xor(rs, 1);
      rs += __shfl_xor(rs, 2);
      rs += __shfl_xor(rs, 4);
      rs += __shfl_xor(rs, 8);
      l_run[v] = l_run[v] * alpha[v] + rs;
      m_run[v] = mn;
      p0[v] = e0v;
      p1[v] = e1v;
    }
    __syncthreads(); /* sred reads done before next iteration's stores */

#pragma unroll
    for (int nt = 0; nt < 12; nt++)
#pragma unroll
      for (int v = 0; v < 8; v++) o[nt][v] *= alpha[v];

    /* ---- C-layout P -> A-layout via per-wave private LDS tile ---- */
    bf16* lpw = lp[w];
#pragma unroll
    for (int v = 0; v < 8; v++) {
      int r = (v + 8 * half) * KP;
      lpw[r + ln]      = (bf16)p0[v];
      lpw[r + 16 + ln] = (bf16)p1[v];
    }
    v16bf Ap = load_frag(&lpw[ln * KP + 8 * half], &lpw[ln * KP + 16 + 8 * half]);

    /* ---- O += P·V over this wave's E slice ---- */
#pragma unroll
    for (int nt = 0; nt < 12; nt++) {
      const bf16* vp = vT + ((size_t)b * E_ + ew + nt * 16 + ln) * (size_t)S_ + t0 + 8 * half;
      v16bf Bv = load_frag(vp, vp + 16);
      o[nt] = __builtin_amdgcn_wmma_f32_16x16x32_bf16(false, Ap, false, Bv, (short)0, o[nt], false, false);
    }
  }

#pragma unroll
  for (int nt = 0; nt < 12; nt++)
#pragma unroll
    for (int v = 0; v < 8; v++) {
      float val = o[nt][v] / l_run[v];
      int row   = s0 + v + 8 * half;
      h[((size_t)(b * S_ + row)) * E_ + ew + nt * 16 + ln] = val;
    }
}

/* ---------------- fused residual add + LayerNorm ---------------- */

__global__ __launch_bounds__(256) void add_ln(
    const float* __restrict__ a, const float* __restrict__ bsrc,
    const float* __restrict__ g, const float* __restrict__ be,
    float* __restrict__ of, bf16* __restrict__ ob)
{
  __shared__ float red[256];
  const int row = blockIdx.x;
  const int t   = threadIdx.x;
  const float* pa = a + (size_t)row * E_;
  const float* pb = bsrc + (size_t)row * E_;

  float vals[3];
  float s = 0.f;
#pragma unroll
  for (int i = 0; i < 3; i++) {
    int c = t + i * 256;
    float v = pa[c] + pb[c];
    vals[i] = v;
    s += v;
  }
  red[t] = s;
  __syncthreads();
  for (int st = 128; st > 0; st >>= 1) {
    if (t < st) red[t] += red[t + st];
    __syncthreads();
  }
  float mean = red[0] * (1.0f / E_);
  __syncthreads();

  float s2 = 0.f;
#pragma unroll
  for (int i = 0; i < 3; i++) {
    float d = vals[i] - mean;
    s2 += d * d;
  }
  red[t] = s2;
  __syncthreads();
  for (int st = 128; st > 0; st >>= 1) {
    if (t < st) red[t] += red[t + st];
    __syncthreads();
  }
  float inv = rsqrtf(red[0] * (1.0f / E_) + 1e-5f);

#pragma unroll
  for (int i = 0; i < 3; i++) {
    int c = t + i * 256;
    float yv = (vals[i] - mean) * inv * g[c] + be[c];
    of[(size_t)row * E_ + c] = yv;
    if (ob) ob[(size_t)row * E_ + c] = (bf16)yv;
  }
}

/* ---------------- host orchestration ---------------- */

extern "C" void kernel_launch(void* const* d_in, const int* in_sizes, int n_in,
                              void* d_out, int out_size, void* d_ws, size_t ws_size,
                              hipStream_t stream)
{
  (void)in_sizes; (void)n_in; (void)out_size; (void)ws_size;
  const float* x   = (const float*)d_in[0];
  const float* Wq  = (const float*)d_in[1];
  const float* bq  = (const float*)d_in[2];
  const float* Wk  = (const float*)d_in[3];
  const float* bk  = (const float*)d_in[4];
  const float* Wv  = (const float*)d_in[5];
  const float* bv  = (const float*)d_in[6];
  const float* g1  = (const float*)d_in[7];
  const float* be1 = (const float*)d_in[8];
  const float* W1  = (const float*)d_in[9];
  const float* b1  = (const float*)d_in[10];
  const float* W2  = (const float*)d_in[11];
  const float* b2  = (const float*)d_in[12];
  const float* g2  = (const float*)d_in[13];
  const float* be2 = (const float*)d_in[14];
  float* out = (float*)d_out;

  char* ws = (char*)d_ws;
  size_t off = 0;
  auto alloc = [&](size_t bytes) -> void* {
    void* p = ws + off;
    off += (bytes + 255) & ~(size_t)255;
    return p;
  };
  bf16* xb   = (bf16*)alloc((size_t)NTOK * E_ * 2);
  bf16* WqT  = (bf16*)alloc((size_t)E_ * E_ * 2);
  bf16* WkT  = (bf16*)alloc((size_t)E_ * E_ * 2);
  bf16* WvT  = (bf16*)alloc((size_t)E_ * E_ * 2);
  bf16* W1T  = (bf16*)alloc((size_t)E_ * H_ * 2);
  bf16* W2T  = (bf16*)alloc((size_t)H_ * E_ * 2);
  bf16* qbuf = (bf16*)alloc((size_t)NTOK * E_ * 2);
  bf16* kbuf = (bf16*)alloc((size_t)NTOK * E_ * 2);
  bf16* vbuf = (bf16*)alloc((size_t)NTOK * E_ * 2);
  bf16* vT   = (bf16*)alloc((size_t)NTOK * E_ * 2);
  bf16* act  = qbuf;   /* reuse q/k/v/vT region (4*25.2MB == NTOK*H*2) after attention */
  float* hbuf = (float*)alloc((size_t)NTOK * E_ * 4);
  float* mlp  = hbuf;  /* h consumed by ln1 before mlp2 writes */
  float* ybuf = (float*)alloc((size_t)NTOK * E_ * 4);
  bf16*  yb   = (bf16*)alloc((size_t)NTOK * E_ * 2);

  const int nx = NTOK * E_;
  conv_f32_bf16<<<(nx + 255) / 256, 256, 0, stream>>>(x, xb, nx);
  transpose_conv<<<(E_ * E_ + 255) / 256, 256, 0, stream>>>(Wq, WqT, E_, E_);
  transpose_conv<<<(E_ * E_ + 255) / 256, 256, 0, stream>>>(Wk, WkT, E_, E_);
  transpose_conv<<<(E_ * E_ + 255) / 256, 256, 0, stream>>>(Wv, WvT, E_, E_);
  transpose_conv<<<(E_ * H_ + 255) / 256, 256, 0, stream>>>(W1, W1T, E_, H_);
  transpose_conv<<<(H_ * E_ + 255) / 256, 256, 0, stream>>>(W2, W2T, H_, E_);

  dim3 gE(E_ / 128, NTOK / 128);
  gemm_bf16_wmma<<<gE, 256, 0, stream>>>(xb, WqT, bq, nullptr, qbuf, NTOK, E_, E_, 0);
  gemm_bf16_wmma<<<gE, 256, 0, stream>>>(xb, WkT, bk, nullptr, kbuf, NTOK, E_, E_, 0);
  gemm_bf16_wmma<<<gE, 256, 0, stream>>>(xb, WvT, bv, nullptr, vbuf, NTOK, E_, E_, 0);

  transpose_v<<<(nx + 255) / 256, 256, 0, stream>>>(vbuf, vT);

  attn_wmma<<<B_ * (S_ / 16), 128, 0, stream>>>(qbuf, kbuf, vT, hbuf);

  add_ln<<<NTOK, 256, 0, stream>>>(x, hbuf, g1, be1, ybuf, yb);

  dim3 gH(H_ / 128, NTOK / 128);
  gemm_bf16_wmma<<<gH, 256, 0, stream>>>(yb, W1T, b1, nullptr, act, NTOK, H_, E_, 1);
  gemm_bf16_wmma<<<gE, 256, 0, stream>>>(act, W2T, b2, mlp, nullptr, NTOK, E_, H_, 1);

  add_ln<<<NTOK, 256, 0, stream>>>(ybuf, mlp, g2, be2, out, nullptr);
}
